// MultichannelMultiheadAttention_68917045231704
// MI455X (gfx1250) — matrix operand order repaired and checked by reading, not compiled, not run on hardware
//
#include <hip/hip_runtime.h>
#include <hip/hip_bf16.h>

// ---------------------------------------------------------------------------
// MultichannelMultiheadAttention for MI455X (gfx1250, wave32, WMMA bf16)
// B=2 C=8 L=512 CH=256 CW=8 NH=8 head_dim=256 ROT=32
// Head feature index is stored PERMUTED: d' = (g&31) + 32*w  (d = (g&31)*8+w).
// Attention is invariant under a consistent permutation of the head dim;
// RoPE pairs (2i,2i+1) map to (d', d'+32); un-permuted on final scatter.
// Attention kernels stage shared tiles in LDS via async global->LDS copies.
// ---------------------------------------------------------------------------

typedef __bf16 bf16;
typedef __attribute__((ext_vector_type(16))) __bf16 v16bf;
typedef __attribute__((ext_vector_type(8)))  __bf16 v8bf;
typedef __attribute__((ext_vector_type(4)))  __bf16 v4bf;
typedef __attribute__((ext_vector_type(8)))  float  v8f;

#define UE 16777216LL   // one "unit" = 16M bf16 elements (32 MiB)

// --- WMMA fragment helpers (16x16x32 bf16, contiguous k = 16h+s in both A and B) -------
// A (16x32): lane holds row (lane&15); K = {16h..16h+15}
__device__ __forceinline__ v16bf load_a_frag(const bf16* __restrict__ row,
                                             int kk, int half) {
  v8bf lo = *(const v8bf*)(row + kk + 16 * half);
  v8bf hi = *(const v8bf*)(row + kk + 16 * half + 8);
  v16bf a;
#pragma unroll
  for (int i = 0; i < 8; ++i) { a[i] = lo[i]; a[8 + i] = hi[i]; }
  return a;
}

// B (32x16): lane holds column (lane&15); element j = B[kbase+j, n],
// kbase = kk + 16*(lane>>4). Source must be 16 contiguous bf16.
__device__ __forceinline__ v16bf load_b_frag_contig(const bf16* __restrict__ p) {
  v8bf lo = *(const v8bf*)(p);
  v8bf hi = *(const v8bf*)(p + 8);
  v16bf b;
#pragma unroll
  for (int i = 0; i < 8; ++i) { b[i] = lo[i]; b[8 + i] = hi[i]; }
  return b;
}

// Async copy of 16 bytes/lane from global to LDS (CDNA5 ASYNCcnt path).
__device__ __forceinline__ void async_g2l_b128(unsigned lds_byte_addr,
                                               const void* gaddr) {
  asm volatile("global_load_async_to_lds_b128 %0, %1, off"
               :: "v"(lds_byte_addr), "v"(gaddr) : "memory");
}
__device__ __forceinline__ void wait_async0() {
  asm volatile("s_wait_asynccnt 0x0" ::: "memory");
}

// ---------------------------------------------------------------------------
// K0: one-time fp32 -> bf16 conversion of the 4 mlin weight matrices.
// ---------------------------------------------------------------------------
__global__ void weights_to_bf16(const float* __restrict__ wq,
                                const float* __restrict__ wk,
                                const float* __restrict__ wv,
                                const float* __restrict__ wo,
                                bf16* __restrict__ bq, bf16* __restrict__ bk,
                                bf16* __restrict__ bv, bf16* __restrict__ bo) {
  int i = blockIdx.x * blockDim.x + threadIdx.x;  // 524288
  bq[i] = (bf16)wq[i]; bk[i] = (bf16)wk[i];
  bv[i] = (bf16)wv[i]; bo[i] = (bf16)wo[i];
}

// ---------------------------------------------------------------------------
// K1: fused 3x3 conv for q/k/v over spatial (CH=256, CW=8), 8 channels.
// y fastest -> lane-contiguous bf16 stores into GEMM-A layout
// z[(c*8+w)*1024 + n][h], n=b*512+l, h=ch row.
// ---------------------------------------------------------------------------
__global__ void conv3_qkv(const float* __restrict__ xin,
                          const float* __restrict__ wq,
                          const float* __restrict__ wk,
                          const float* __restrict__ wv,
                          bf16* __restrict__ zq,
                          bf16* __restrict__ zk,
                          bf16* __restrict__ zv) {
  int idx = blockIdx.x * blockDim.x + threadIdx.x;  // 16,777,216 total
  int y  = idx & 255;
  int xw = (idx >> 8) & 7;
  int c  = (idx >> 11) & 7;
  int n  = idx >> 14;                 // 0..1023
  int b  = n >> 9, l = n & 511;
  float sq = 0.f, sk = 0.f, sv = 0.f;
#pragma unroll
  for (int ci = 0; ci < 8; ++ci) {
    const float* src = xin + ((((long long)b * 8 + ci) * 512 + l) * 256) * 8;
    const float* qW = wq + (c * 8 + ci) * 9;
    const float* kW = wk + (c * 8 + ci) * 9;
    const float* vW = wv + (c * 8 + ci) * 9;
#pragma unroll
    for (int ky = 0; ky < 3; ++ky) {
      int yy = y + ky - 1;
      if (yy < 0 || yy > 255) continue;
#pragma unroll
      for (int kx = 0; kx < 3; ++kx) {
        int xx = xw + kx - 1;
        if (xx < 0 || xx > 7) continue;
        float v = src[yy * 8 + xx];
        int wi = ky * 3 + kx;
        sq = fmaf(v, qW[wi], sq);
        sk = fmaf(v, kW[wi], sk);
        sv = fmaf(v, vW[wi], sv);
      }
    }
  }
  long long o = (((long long)(c * 8 + xw)) * 1024 + n) * 256 + y;
  zq[o] = (bf16)sq; zk[o] = (bf16)sk; zv[o] = (bf16)sv;
}

// ---------------------------------------------------------------------------
// K2: mlin as 64 GEMMs (M=1024,N=256,K=256), register-blocked N x4.
// Stores permuted head layout Q[hd][l][d'], d' = (g&31) + 32*w.
// ---------------------------------------------------------------------------
__global__ void mlin_heads(const bf16* __restrict__ Z,
                           const bf16* __restrict__ Wb,
                           bf16* __restrict__ Qout) {
  int lane = threadIdx.x & 31;
  int wid  = blockIdx.x * (blockDim.x >> 5) + (threadIdx.x >> 5);  // 16384
  int cw   = wid >> 8;          // 0..63 -> (c,w)
  int t    = wid & 255;
  int tm   = t & 63;            // M tile (1024/16)
  int tnb  = t >> 6;            // N block of 64 cols (0..3)
  int c = cw >> 3, w = cw & 7;
  int half = lane >> 4, l15 = lane & 15;
  const bf16* arow  = Z + (((long long)(c * 8 + w)) * 1024 + tm * 16 + l15) * 256;
  int g0 = tnb * 64;
  const bf16* brow0 = Wb + ((long long)c * 256 + g0 + l15) * 256;
  v8f acc[4] = {};
#pragma unroll
  for (int kk = 0; kk < 256; kk += 32) {
    __builtin_prefetch(arow + kk + 64, 0, 1);
    v16bf a = load_a_frag(arow, kk, half);
    int kb = kk + 16 * half;
#pragma unroll
    for (int j = 0; j < 4; ++j) {
      v16bf b = load_b_frag_contig(brow0 + j * (16 * 256) + kb);
      acc[j] = __builtin_amdgcn_wmma_f32_16x16x32_bf16(false, a, false, b,
                                                       (short)0, acc[j],
                                                       false, false);
    }
  }
#pragma unroll
  for (int j = 0; j < 4; ++j) {
    int g  = g0 + j * 16 + l15;
    int nh = g >> 5;
    int dp = (g & 31) + 32 * w;     // permuted head index
#pragma unroll
    for (int v = 0; v < 8; ++v) {
      int nrow = tm * 16 + v + 8 * half;
      int b_ = nrow >> 9, l = nrow & 511;
      long long o = ((((long long)(b_ * 8 + c) * 8 + nh) * 512 + l) * 256 + dp);
      Qout[o] = (bf16)acc[j][v];
    }
  }
}

// ---------------------------------------------------------------------------
// K3: RoPE in-place. Original pair (2i,2i+1) lives at permuted offsets
// (d', d'+32) with d' = (2i>>3) + 32*(2i&7).
// ---------------------------------------------------------------------------
__global__ void rope_qk(bf16* __restrict__ Q, bf16* __restrict__ K) {
  int idx = blockIdx.x * blockDim.x + threadIdx.x;  // 2*128*512*16
  int i  = idx & 15;
  int l  = (idx >> 4) & 511;
  int hd = (idx >> 13) & 127;
  int which = idx >> 20;
  bf16* base = (which ? K : Q) + ((long long)hd * 512 + l) * 256;
  int d0  = 2 * i;
  int dp0 = (d0 >> 3) + 32 * (d0 & 7);
  float inv = __expf(-(float)(2 * i) * (9.210340371976184f / 32.0f));
  float ang = (float)l * inv;
  float cs = __cosf(ang), sn = __sinf(ang);
  float x0 = (float)base[dp0], x1 = (float)base[dp0 + 32];
  base[dp0]      = (bf16)(x0 * cs - x1 * sn);
  base[dp0 + 32] = (bf16)(x1 * cs + x0 * sn);
}

// ---------------------------------------------------------------------------
// K4a: S = Q K^T / 16 + prev_qk per head. Block = 128M x 64N; the 64x32 bf16
// K-tile (4 KB) is async-staged to LDS once per K-step and shared by all 8
// waves; B-frags come from LDS (ds_load_b128), A-frags from global.
// LDS rows padded to 40 bf16 (20-dword stride: conflict-free, 16B-aligned).
// ---------------------------------------------------------------------------
__global__ void attn_qk(const bf16* __restrict__ Q, const bf16* __restrict__ K,
                        const float* __restrict__ prevqk,
                        float* __restrict__ qkout) {
  __shared__ bf16 tB[64][40];
  int tid  = threadIdx.x;
  int lane = tid & 31, wv = tid >> 5;       // wave 0..7
  int bid  = blockIdx.x;                    // 4096
  int hd   = bid >> 5;                      // 0..127
  int tb   = bid & 31;
  int mb   = tb & 3;                        // M block of 128 rows
  int nb   = tb >> 2;                       // N block of 64 cols
  int half = lane >> 4, l15 = lane & 15;
  int tm   = mb * 8 + wv;                   // wave's M tile (0..31)
  const bf16* arow = Q + ((long long)hd * 512 + tm * 16 + l15) * 256;
  int n0 = nb * 64;
  const bf16* ksrc = K + ((long long)hd * 512 + n0 + (tid >> 2)) * 256 +
                     (tid & 3) * 8;
  unsigned ldst = (unsigned)(size_t)&tB[tid >> 2][(tid & 3) * 8];
  v8f acc[4] = {};
  for (int kk = 0; kk < 256; kk += 32) {
    __syncthreads();                        // previous step's reads done
    async_g2l_b128(ldst, ksrc + kk);        // stage 64x32 K-tile
    wait_async0();
    __syncthreads();                        // whole tile visible
    __builtin_prefetch(arow + kk + 64, 0, 1);
    v16bf a = load_a_frag(arow, kk, half);
#pragma unroll
    for (int j = 0; j < 4; ++j) {
      v16bf b = load_b_frag_contig(&tB[j * 16 + l15][16 * half]);
      acc[j] = __builtin_amdgcn_wmma_f32_16x16x32_bf16(false, a, false, b,
                                                       (short)0, acc[j],
                                                       false, false);
    }
  }
#pragma unroll
  for (int j = 0; j < 4; ++j) {
    int mcol = n0 + j * 16 + l15;
#pragma unroll
    for (int v = 0; v < 8; ++v) {
      int lrow = tm * 16 + v + 8 * half;
      long long o = ((long long)hd * 512 + lrow) * 512 + mcol;
      qkout[o] = acc[j][v] * 0.0625f + prevqk[o];
    }
  }
}

// ---------------------------------------------------------------------------
// K4b: row softmax over 512 columns; one wave32 per row; bf16 P out.
// ---------------------------------------------------------------------------
__global__ void softmax_rows(const float* __restrict__ qk,
                             bf16* __restrict__ P) {
  int lane = threadIdx.x & 31;
  int row  = blockIdx.x * (blockDim.x >> 5) + (threadIdx.x >> 5);  // 65536
  const float* r = qk + (long long)row * 512;
  float vals[16];
  float mx = -3.4e38f;
#pragma unroll
  for (int j = 0; j < 16; ++j) {
    vals[j] = r[lane + 32 * j];
    mx = fmaxf(mx, vals[j]);
  }
#pragma unroll
  for (int m = 16; m >= 1; m >>= 1) mx = fmaxf(mx, __shfl_xor(mx, m, 32));
  float s = 0.f;
#pragma unroll
  for (int j = 0; j < 16; ++j) { vals[j] = __expf(vals[j] - mx); s += vals[j]; }
#pragma unroll
  for (int m = 16; m >= 1; m >>= 1) s += __shfl_xor(s, m, 32);
  float inv = 1.f / s;
  bf16* pr = P + (long long)row * 512;
#pragma unroll
  for (int j = 0; j < 16; ++j) pr[lane + 32 * j] = (bf16)(vals[j] * inv);
}

// ---------------------------------------------------------------------------
// K4t: V[hd][m][d'] -> Vt[hd][d'][m] via LDS 32x32 tiles.
// ---------------------------------------------------------------------------
__global__ void transpose_v(const bf16* __restrict__ V,
                            bf16* __restrict__ Vt) {
  __shared__ bf16 tile[32][36];
  int bid = blockIdx.x;                 // 128 heads * 16 mt * 8 dt = 16384
  int dt = bid & 7;
  int mt = (bid >> 3) & 15;
  int hd = bid >> 7;
  int t  = threadIdx.x;
  int r  = t >> 3;                      // 0..31
  int cg = (t & 7) * 4;                 // 0,4,...,28
  v4bf in = *(const v4bf*)(V + ((long long)hd * 512 + mt * 32 + r) * 256 +
                           dt * 32 + cg);
#pragma unroll
  for (int j = 0; j < 4; ++j) tile[r][cg + j] = in[j];
  __syncthreads();
  v4bf outv;
#pragma unroll
  for (int j = 0; j < 4; ++j) outv[j] = tile[cg + j][r];
  *(v4bf*)(Vt + ((long long)hd * 256 + dt * 32 + r) * 512 + mt * 32 + cg) = outv;
}

// ---------------------------------------------------------------------------
// K4c: A = P V per head (512x256x512). Block = 64M x 128N; the 64x32 bf16
// P-tile is async-staged to LDS once per K-step and shared by all 8 waves;
// A-frags come from LDS, B-frags (Vt rows) from global. Output un-permutes
// d' into Aout[(c*8+w)*1024+n][h], w=d'>>5, h=nh*32+(d'&31).
// ---------------------------------------------------------------------------
__global__ void attn_av(const bf16* __restrict__ P, const bf16* __restrict__ Vt,
                        bf16* __restrict__ Aout) {
  __shared__ bf16 tA[64][40];
  int tid  = threadIdx.x;
  int lane = tid & 31, wv = tid >> 5;
  int bid  = blockIdx.x;                // 2048
  int hd   = bid >> 4;                  // 0..127
  int tb   = bid & 15;
  int tmb  = tb >> 1;                   // M block of 64 rows (0..7)
  int nbh  = tb & 1;                    // N half (8 tiles each)
  int half = lane >> 4, l15 = lane & 15;
  int tn   = nbh * 8 + wv;              // N tile (0..15)
  int dcol = tn * 16 + l15;
  const bf16* btrow = Vt + ((long long)hd * 256 + dcol) * 512;
  const bf16* psrc  = P + ((long long)hd * 512 + tmb * 64 + (tid >> 2)) * 512 +
                      (tid & 3) * 8;
  unsigned ldst = (unsigned)(size_t)&tA[tid >> 2][(tid & 3) * 8];
  v8f acc[4] = {};
  for (int kk = 0; kk < 512; kk += 32) {
    __syncthreads();
    async_g2l_b128(ldst, psrc + kk);      // stage 64x32 P-tile
    wait_async0();
    __syncthreads();
    __builtin_prefetch(btrow + kk + 64, 0, 1);
    v16bf b = load_b_frag_contig(btrow + kk + 16 * half);
#pragma unroll
    for (int i = 0; i < 4; ++i) {
      v16bf a = load_a_frag(&tA[i * 16 + l15][0], 0, half);
      acc[i] = __builtin_amdgcn_wmma_f32_16x16x32_bf16(false, a, false, b,
                                                       (short)0, acc[i],
                                                       false, false);
    }
  }
  int nh = hd & 7, c = (hd >> 3) & 7, b_ = hd >> 6;
  int w = dcol >> 5;                    // un-permute: d = (d'&31)*8 + (d'>>5)
  int h = (nh << 5) + (dcol & 31);
#pragma unroll
  for (int i = 0; i < 4; ++i) {
#pragma unroll
    for (int v = 0; v < 8; ++v) {
      int lrow = tmb * 64 + i * 16 + v + 8 * half;
      int n = b_ * 512 + lrow;
      long long o = (((long long)(c * 8 + w)) * 1024 + n) * 256 + h;
      Aout[o] = (bf16)acc[i][v];
    }
  }
}

// ---------------------------------------------------------------------------
// K5: final mlin with wo (bf16), N-blocked x4; fp32 out[b][c][l][g][w].
// ---------------------------------------------------------------------------
__global__ void mlin_out(const bf16* __restrict__ A,
                         const bf16* __restrict__ Wb,
                         float* __restrict__ outp) {
  int lane = threadIdx.x & 31;
  int wid  = blockIdx.x * (blockDim.x >> 5) + (threadIdx.x >> 5);  // 16384
  int cw   = wid >> 8;
  int t    = wid & 255;
  int tm   = t & 63;
  int tnb  = t >> 6;
  int c = cw >> 3, w = cw & 7;
  int half = lane >> 4, l15 = lane & 15;
  const bf16* arow  = A + (((long long)(c * 8 + w)) * 1024 + tm * 16 + l15) * 256;
  int g0 = tnb * 64;
  const bf16* brow0 = Wb + ((long long)c * 256 + g0 + l15) * 256;
  v8f acc[4] = {};
#pragma unroll
  for (int kk = 0; kk < 256; kk += 32) {
    __builtin_prefetch(arow + kk + 64, 0, 1);
    v16bf a = load_a_frag(arow, kk, half);
    int kb = kk + 16 * half;
#pragma unroll
    for (int j = 0; j < 4; ++j) {
      v16bf b = load_b_frag_contig(brow0 + j * (16 * 256) + kb);
      acc[j] = __builtin_amdgcn_wmma_f32_16x16x32_bf16(false, a, false, b,
                                                       (short)0, acc[j],
                                                       false, false);
    }
  }
#pragma unroll
  for (int j = 0; j < 4; ++j) {
    int g = g0 + j * 16 + l15;
#pragma unroll
    for (int v = 0; v < 8; ++v) {
      int nrow = tm * 16 + v + 8 * half;
      int b_ = nrow >> 9, l = nrow & 511;
      long long o = ((((long long)(b_ * 8 + c) * 512 + l) * 256 + g)) * 8 + w;
      outp[o] = acc[j][v];
    }
  }
}

// ---------------------------------------------------------------------------
extern "C" void kernel_launch(void* const* d_in, const int* in_sizes, int n_in,
                              void* d_out, int out_size, void* d_ws,
                              size_t ws_size, hipStream_t stream) {
  (void)in_sizes; (void)n_in; (void)out_size; (void)ws_size;
  const float* x       = (const float*)d_in[0];
  const float* prev_qk = (const float*)d_in[1];
  const float* cqw     = (const float*)d_in[2];
  const float* ckw     = (const float*)d_in[3];
  const float* cvw     = (const float*)d_in[4];
  const float* wq      = (const float*)d_in[5];
  const float* wk      = (const float*)d_in[6];
  const float* wv      = (const float*)d_in[7];
  const float* wo      = (const float*)d_in[8];

  float* outp  = (float*)d_out;            // (2,8,512,256,8) = 16,777,216 f32
  float* qkout = outp + 16777216;          // (2,8,8,512,512) = 33,554,432 f32

  bf16* wsb = (bf16*)d_ws;                 // bf16 workspace, units of UE elems
  bf16 *zq = wsb,          *zk = wsb + UE,     *zv = wsb + 2 * UE;
  bf16 *Q  = wsb + 3 * UE, *Kb = wsb + 4 * UE, *Vb = wsb + 5 * UE;
  bf16 *Vt = wsb + 6 * UE;
  bf16 *bwq = wsb + 7 * UE,       *bwk = bwq + 524288,
       *bwv = bwk + 524288,       *bwo = bwv + 524288;
  bf16 *P    = wsb;                        // reuses zq/zk (dead after mlin)
  bf16 *Aout = wsb + 3 * UE;               // reuses Q (dead after attn_qk)

  weights_to_bf16<<< 2048, 256, 0, stream>>>(wq, wk, wv, wo, bwq, bwk, bwv, bwo);
  conv3_qkv      <<<65536, 256, 0, stream>>>(x, cqw, ckw, cvw, zq, zk, zv);
  mlin_heads     <<< 2048, 256, 0, stream>>>(zq, bwq, Q);
  mlin_heads     <<< 2048, 256, 0, stream>>>(zk, bwk, Kb);
  mlin_heads     <<< 2048, 256, 0, stream>>>(zv, bwv, Vb);
  rope_qk        <<< 8192, 256, 0, stream>>>(Q, Kb);
  transpose_v    <<<16384, 256, 0, stream>>>(Vb, Vt);
  attn_qk        <<< 4096, 256, 0, stream>>>(Q, Kb, prev_qk, qkout);
  softmax_rows   <<< 8192, 256, 0, stream>>>(qkout, P);
  attn_av        <<< 2048, 256, 0, stream>>>(P, Vt, Aout);
  mlin_out       <<< 2048, 256, 0, stream>>>(Aout, bwo, outp);
}
